// TransformerGenerator_25769803776214
// MI455X (gfx1250) — matrix operand order repaired and checked by reading, not compiled, not run on hardware
//
#include <hip/hip_runtime.h>
#include <math.h>

// ---------------- problem constants ----------------
#define BATCH   8
#define EDIM    1536
#define HDIM    256
#define NHEAD   8
#define HEADD   32
#define SEQL    50
#define VOCAB   50257
#define NLAYER  4
#define FFDIM   2048
#define NSTEP   49            // L-1 decode steps
#define BOSTOK  50256

// ---------------- vector types for WMMA ----------------
typedef __attribute__((ext_vector_type(16))) __bf16 v16bf;
typedef __attribute__((ext_vector_type(8)))  __bf16 bf16x8;
typedef __attribute__((ext_vector_type(8)))  float  v8f;

__device__ __forceinline__ unsigned short f2bfu(float f) {
    union { float f; unsigned int u; } c; c.f = f;
    unsigned int u = c.u;
    u += 0x7fffu + ((u >> 16) & 1u);      // round-to-nearest-even
    return (unsigned short)(u >> 16);
}

__device__ __forceinline__ v16bf cat8(bf16x8 lo, bf16x8 hi) {
    return __builtin_shufflevector(lo, hi, 0,1,2,3,4,5,6,7,8,9,10,11,12,13,14,15);
}

__device__ __forceinline__ v16bf zero16() {
    bf16x8 z;
#pragma unroll
    for (int i = 0; i < 8; ++i) z[i] = (__bf16)0.0f;
    return cat8(z, z);
}

// ---------------- fp32 -> bf16 convert ----------------
__global__ void cvt_bf16_k(const float* __restrict__ s, unsigned short* __restrict__ d, int n) {
    int i = blockIdx.x * blockDim.x + threadIdx.x;
    int stride = gridDim.x * blockDim.x;
    for (; i < n; i += stride) d[i] = f2bfu(s[i]);
}

// ---------------- generic WMMA GEMM: C[M,N] = A[M,K] * W[N,K]^T (+bias)(+relu)(+resid) ----------------
// One 16x16 output tile per wave32; bf16 inputs, fp32 accumulate.
__global__ void __launch_bounds__(128) wmma_gemm_k(
    const unsigned short* __restrict__ A,    // [M,K] bf16
    const unsigned short* __restrict__ W,    // [N,K] bf16 (row-major, PyTorch weight layout)
    const float* __restrict__ bias,          // [N] or null
    const float* __restrict__ resid,         // [M,N] or null
    float* __restrict__ C,                   // [M,N]
    int M, int N, int K, int relu)
{
    const int lane = threadIdx.x & 31;
    const int wave = threadIdx.x >> 5;
    const int tilesN = N >> 4;
    const int totalTiles = (M >> 4) * tilesN;
    const int tile = blockIdx.x * 4 + wave;
    if (tile >= totalTiles) return;                 // uniform per wave -> EXEC stays full
    const int tm = tile / tilesN, tn = tile % tilesN;
    const int r16 = lane & 15;
    const int akb = (lane >> 4) << 3;               // A: K octet base 0/8 (ISA 16-bit A 16x32 layout)
    const int bkb = (lane >> 4) << 3;               // B: K octet base 0/8 (same K order as A)
    const size_t arow = (size_t)(tm * 16 + r16) * (size_t)K;
    const size_t brow = (size_t)(tn * 16 + r16) * (size_t)K;

    v8f acc = {};
    for (int k0 = 0; k0 < K; k0 += 32) {
        const bf16x8* ap = (const bf16x8*)(A + arow + k0 + akb);
        const bf16x8* bp = (const bf16x8*)(W + brow + k0 + bkb);
        v16bf av = cat8(ap[0], ap[2]);              // K = akb..akb+7 , akb+16..akb+23
        v16bf bv = cat8(bp[0], bp[2]);              // K = bkb..bkb+7 , bkb+16..bkb+23
        acc = __builtin_amdgcn_wmma_f32_16x16x32_bf16(
                  false, av, false, bv, (short)0, acc, false, false);
    }

    const int ocol = tn * 16 + r16;
    const int rbase = tm * 16 + ((lane >> 4) << 3);
    const float bcol = bias ? bias[ocol] : 0.0f;
#pragma unroll
    for (int r = 0; r < 8; ++r) {
        const int orow = rbase + r;
        float v = acc[r] + bcol;
        if (relu) v = fmaxf(v, 0.0f);
        if (resid) v += resid[(size_t)orow * N + ocol];
        C[(size_t)orow * N + ocol] = v;
    }
}

// ---------------- vocab projection at position t: logits[b, t, :] ----------------
__global__ void __launch_bounds__(128) wmma_logits_k(
    const unsigned short* __restrict__ Xbf,  // [B*SEQL, HDIM] bf16 (final x)
    const unsigned short* __restrict__ Wfc,  // [VOCAB, HDIM] bf16
    const float* __restrict__ fcb,           // [VOCAB]
    float* __restrict__ out,                 // [B, NSTEP, VOCAB]
    int t)
{
    const int tilesN = (VOCAB + 15) >> 4;            // 3142
    const int lane = threadIdx.x & 31;
    const int wave = threadIdx.x >> 5;
    const int tn = blockIdx.x * 4 + wave;
    if (tn >= tilesN) return;
    const int r16 = lane & 15;
    const int akb = (lane >> 4) << 3;
    const int bkb = (lane >> 4) << 3;
    const int b = r16;                               // A row = batch index (rows 8..15 padded)
    const bool avalid = (b < BATCH);
    const int col = tn * 16 + r16;
    const bool bvalid = (col < VOCAB);
    const size_t arow = ((size_t)b * SEQL + t) * HDIM;
    const size_t brow = (size_t)col * HDIM;

    v8f acc = {};
    for (int k0 = 0; k0 < HDIM; k0 += 32) {
        v16bf av = zero16(), bv = zero16();
        if (avalid) {
            const bf16x8* ap = (const bf16x8*)(Xbf + arow + k0 + akb);
            av = cat8(ap[0], ap[2]);
        }
        if (bvalid) {
            const bf16x8* bp = (const bf16x8*)(Wfc + brow + k0 + bkb);
            bv = cat8(bp[0], bp[2]);
        }
        acc = __builtin_amdgcn_wmma_f32_16x16x32_bf16(
                  false, av, false, bv, (short)0, acc, false, false);
    }
    if (bvalid) {
        const float bb = fcb[col];
#pragma unroll
        for (int r = 0; r < 8; ++r) {
            const int ob = ((lane >> 4) << 3) + r;
            if (ob < BATCH)
                out[((size_t)ob * NSTEP + t) * VOCAB + col] = acc[r] + bb;
        }
    }
}

// ---------------- causal self-attention (one block per (batch, head)) ----------------
__global__ void self_attn_k(const float* __restrict__ qkv,   // [B*SEQL, 3*HDIM]
                            float* __restrict__ att)          // [B*SEQL, HDIM]
{
    const int b = blockIdx.x >> 3;
    const int h = blockIdx.x & 7;
    __shared__ float qs[SEQL][HEADD], ks[SEQL][HEADD], vs[SEQL][HEADD];
    for (int idx = threadIdx.x; idx < SEQL * HEADD; idx += blockDim.x) {
        const int l = idx >> 5, d = idx & 31;
        const size_t base = ((size_t)(b * SEQL + l)) * (3 * HDIM) + h * HEADD + d;
        qs[l][d] = qkv[base];
        ks[l][d] = qkv[base + HDIM];
        vs[l][d] = qkv[base + 2 * HDIM];
    }
    __syncthreads();
    const int i = threadIdx.x;
    if (i < SEQL) {
        float sc[SEQL];
        float mx = -3.4e38f;
        const float scale = 0.17677669529663687f;   // 1/sqrt(32)
        for (int j = 0; j <= i; ++j) {
            float s = 0.0f;
            for (int d = 0; d < HEADD; ++d) s += qs[i][d] * ks[j][d];
            s *= scale;
            sc[j] = s;
            mx = fmaxf(mx, s);
        }
        float sum = 0.0f;
        for (int j = 0; j <= i; ++j) { float e = expf(sc[j] - mx); sc[j] = e; sum += e; }
        const float inv = 1.0f / sum;
        for (int d = 0; d < HEADD; ++d) {
            float o = 0.0f;
            for (int j = 0; j <= i; ++j) o += sc[j] * vs[j][d];
            att[(size_t)(b * SEQL + i) * HDIM + h * HEADD + d] = o * inv;
        }
    }
}

// ---------------- LayerNorm over H=256 (one block per row); add_mode: 0 none, 2 per-batch ----------------
__global__ void ln_k(const float* __restrict__ in, const float* __restrict__ add, int add_mode,
                     const float* __restrict__ g, const float* __restrict__ beta,
                     float* __restrict__ out)
{
    __shared__ float red[HDIM];
    const int row = blockIdx.x, tid = threadIdx.x;
    float v = in[(size_t)row * HDIM + tid];
    if (add_mode == 2) v += add[(size_t)(row / SEQL) * HDIM + tid];
    red[tid] = v; __syncthreads();
    for (int s = HDIM / 2; s > 0; s >>= 1) { if (tid < s) red[tid] += red[tid + s]; __syncthreads(); }
    const float m = red[0] * (1.0f / HDIM); __syncthreads();
    const float d = v - m;
    red[tid] = d * d; __syncthreads();
    for (int s = HDIM / 2; s > 0; s >>= 1) { if (tid < s) red[tid] += red[tid + s]; __syncthreads(); }
    const float var = red[0] * (1.0f / HDIM);
    out[(size_t)row * HDIM + tid] = d * rsqrtf(var + 1e-5f) * g[tid] + beta[tid];
}

// ---------------- token embedding + positional encoding ----------------
__global__ void embed_k(const int* __restrict__ tokens, const float* __restrict__ tok_emb,
                        const float* __restrict__ pos_enc, float* __restrict__ x)
{
    const int row = blockIdx.x;            // 0..399
    const int hcol = threadIdx.x;          // 0..255
    const int b = row / SEQL, l = row % SEQL;
    const int tok = tokens[b * SEQL + l];
    x[(size_t)row * HDIM + hcol] = tok_emb[(size_t)tok * HDIM + hcol] + pos_enc[(size_t)l * HDIM + hcol];
}

// ---------------- argmax of logits[b, t, :] with first-max tie-break ----------------
__global__ void argmax_k(const float* __restrict__ logits_all, int t, int* __restrict__ tokens)
{
    const int b = blockIdx.x, tid = threadIdx.x;
    const float* lg = logits_all + ((size_t)b * NSTEP + t) * VOCAB;
    float bv = -3.4e38f; int bi = 0x7fffffff;
    for (int v = tid; v < VOCAB; v += 256) {
        const float f = lg[v];
        if (f > bv) { bv = f; bi = v; }
    }
    __shared__ float sv[256]; __shared__ int si[256];
    sv[tid] = bv; si[tid] = bi; __syncthreads();
    for (int s = 128; s > 0; s >>= 1) {
        if (tid < s) {
            if (sv[tid + s] > sv[tid] || (sv[tid + s] == sv[tid] && si[tid + s] < si[tid])) {
                sv[tid] = sv[tid + s]; si[tid] = si[tid + s];
            }
        }
        __syncthreads();
    }
    if (tid == 0) tokens[b * SEQL + t + 1] = si[0];
}

// ---------------- one-time small projections (fp32, latency-irrelevant) ----------------
__global__ void memproj_k(const float* __restrict__ emb, const float* __restrict__ w,
                          const float* __restrict__ bias, float* __restrict__ memb)
{
    const int b = blockIdx.x, h = threadIdx.x;      // 8 x 256
    float s = bias[h];
    const float* e = emb + (size_t)b * EDIM;
    const float* wr = w + (size_t)h * EDIM;
    for (int k = 0; k < EDIM; ++k) s += e[k] * wr[k];
    memb[b * HDIM + h] = s;
}

// v_l[b,h] = mem[b] . ca_in_w[l][2H+h] + ca_in_b[l][2H+h]   (softmax over 1 key == 1 -> o = v)
__global__ void cav_k(const float* __restrict__ memb, const float* __restrict__ ca_in_w,
                      const float* __restrict__ ca_in_b, float* __restrict__ vtmp)
{
    const int l = blockIdx.x >> 3, b = blockIdx.x & 7, h = threadIdx.x;
    const float* wr = ca_in_w + ((size_t)l * 3 * HDIM + 2 * HDIM + h) * HDIM;
    float s = ca_in_b[l * 3 * HDIM + 2 * HDIM + h];
    const float* m = memb + b * HDIM;
    for (int k = 0; k < HDIM; ++k) s += m[k] * wr[k];
    vtmp[(size_t)(l * BATCH + b) * HDIM + h] = s;
}

// ca_add[l,b,h] = v_l[b] . ca_out_w[l][h] + ca_out_b[l][h]
__global__ void caadd_k(const float* __restrict__ vtmp, const float* __restrict__ ca_out_w,
                        const float* __restrict__ ca_out_b, float* __restrict__ ca_add)
{
    const int l = blockIdx.x >> 3, b = blockIdx.x & 7, h = threadIdx.x;
    const float* wr = ca_out_w + ((size_t)l * HDIM + h) * HDIM;
    float s = ca_out_b[l * HDIM + h];
    const float* v = vtmp + (size_t)(l * BATCH + b) * HDIM;
    for (int k = 0; k < HDIM; ++k) s += v[k] * wr[k];
    ca_add[(size_t)(l * BATCH + b) * HDIM + h] = s;
}

__global__ void init_tokens_k(int* __restrict__ tokens)
{
    const int i = blockIdx.x * blockDim.x + threadIdx.x;
    if (i < BATCH * SEQL) tokens[i] = BOSTOK;
}

// ---------------- host launcher ----------------
extern "C" void kernel_launch(void* const* d_in, const int* in_sizes, int n_in,
                              void* d_out, int out_size, void* d_ws, size_t ws_size,
                              hipStream_t stream)
{
    const float* embedding = (const float*)d_in[0];
    const float* w_proj    = (const float*)d_in[1];
    const float* b_proj    = (const float*)d_in[2];
    const float* tok_emb   = (const float*)d_in[3];
    const float* pos_enc   = (const float*)d_in[4];
    const float* sa_in_w   = (const float*)d_in[5];
    const float* sa_in_b   = (const float*)d_in[6];
    const float* sa_out_w  = (const float*)d_in[7];
    const float* sa_out_b  = (const float*)d_in[8];
    const float* ca_in_w   = (const float*)d_in[9];
    const float* ca_in_b   = (const float*)d_in[10];
    const float* ca_out_w  = (const float*)d_in[11];
    const float* ca_out_b  = (const float*)d_in[12];
    const float* ff1_w     = (const float*)d_in[13];
    const float* ff1_b     = (const float*)d_in[14];
    const float* ff2_w     = (const float*)d_in[15];
    const float* ff2_b     = (const float*)d_in[16];
    const float* ln_g      = (const float*)d_in[17];
    const float* ln_b      = (const float*)d_in[18];
    const float* fc_w      = (const float*)d_in[19];
    const float* fc_b      = (const float*)d_in[20];
    float* out = (float*)d_out;

    // -------- workspace layout --------
    char* ws = (char*)d_ws;
    size_t off = 0;
    auto take = [&](size_t bytes) { size_t o = off; off = (off + bytes + 255) & ~(size_t)255; return o; };

    const size_t N_WSA_IN  = (size_t)NLAYER * 3 * HDIM * HDIM;   // 786432
    const size_t N_WSA_OUT = (size_t)NLAYER * HDIM * HDIM;       // 262144
    const size_t N_WFF1    = (size_t)NLAYER * FFDIM * HDIM;      // 2097152
    const size_t N_WFF2    = (size_t)NLAYER * HDIM * FFDIM;      // 2097152
    const size_t N_WFC     = (size_t)VOCAB * HDIM;               // 12865792
    const size_t ROWS      = (size_t)BATCH * SEQL;               // 400

    unsigned short* wb_sa_in  = (unsigned short*)(ws + take(N_WSA_IN  * 2));
    unsigned short* wb_sa_out = (unsigned short*)(ws + take(N_WSA_OUT * 2));
    unsigned short* wb_ff1    = (unsigned short*)(ws + take(N_WFF1    * 2));
    unsigned short* wb_ff2    = (unsigned short*)(ws + take(N_WFF2    * 2));
    unsigned short* wb_fc     = (unsigned short*)(ws + take(N_WFC     * 2));
    unsigned short* xbf       = (unsigned short*)(ws + take(ROWS * FFDIM * 2));
    float* x    = (float*)(ws + take(ROWS * HDIM  * 4));
    float* tmp  = (float*)(ws + take(ROWS * HDIM  * 4));
    float* att  = (float*)(ws + take(ROWS * HDIM  * 4));
    float* qkv  = (float*)(ws + take(ROWS * 3 * HDIM * 4));
    float* ffh  = (float*)(ws + take(ROWS * FFDIM * 4));
    float* memb = (float*)(ws + take((size_t)BATCH * HDIM * 4));
    float* vtmp = (float*)(ws + take((size_t)NLAYER * BATCH * HDIM * 4));
    float* cadd = (float*)(ws + take((size_t)NLAYER * BATCH * HDIM * 4));
    int* tokens = (int*)(ws + take(ROWS * 4));

    auto cvt = [&](const float* s, unsigned short* d, size_t n) {
        int blocks = (int)((n + 255) / 256);
        if (blocks > 8192) blocks = 8192;
        cvt_bf16_k<<<blocks, 256, 0, stream>>>(s, d, (int)n);
    };

    // -------- one-time setup (every call; deterministic) --------
    cvt(sa_in_w,  wb_sa_in,  N_WSA_IN);
    cvt(sa_out_w, wb_sa_out, N_WSA_OUT);
    cvt(ff1_w,    wb_ff1,    N_WFF1);
    cvt(ff2_w,    wb_ff2,    N_WFF2);
    cvt(fc_w,     wb_fc,     N_WFC);
    memproj_k<<<BATCH, HDIM, 0, stream>>>(embedding, w_proj, b_proj, memb);
    cav_k<<<NLAYER * BATCH, HDIM, 0, stream>>>(memb, ca_in_w, ca_in_b, vtmp);
    caadd_k<<<NLAYER * BATCH, HDIM, 0, stream>>>(vtmp, ca_out_w, ca_out_b, cadd);
    init_tokens_k<<<2, 256, 0, stream>>>(tokens);

    const int logitsTiles = (VOCAB + 15) / 16;                  // 3142
    const int logitsGrid  = (logitsTiles + 3) / 4;              // 786

    // -------- autoregressive decode: 49 strictly sequential steps --------
    for (int t = 0; t < NSTEP; ++t) {
        embed_k<<<(int)ROWS, HDIM, 0, stream>>>(tokens, tok_emb, pos_enc, x);

        for (int l = 0; l < NLAYER; ++l) {
            // ---- self-attention ----
            cvt(x, xbf, ROWS * HDIM);
            wmma_gemm_k<<<300, 128, 0, stream>>>(xbf, wb_sa_in + (size_t)l * 3 * HDIM * HDIM,
                                                 sa_in_b + l * 3 * HDIM, nullptr, qkv,
                                                 (int)ROWS, 3 * HDIM, HDIM, 0);
            self_attn_k<<<BATCH * NHEAD, 128, 0, stream>>>(qkv, att);
            cvt(att, xbf, ROWS * HDIM);
            wmma_gemm_k<<<100, 128, 0, stream>>>(xbf, wb_sa_out + (size_t)l * HDIM * HDIM,
                                                 sa_out_b + l * HDIM, x, tmp,
                                                 (int)ROWS, HDIM, HDIM, 0);
            ln_k<<<(int)ROWS, HDIM, 0, stream>>>(tmp, nullptr, 0,
                                                 ln_g + (l * 3 + 0) * HDIM, ln_b + (l * 3 + 0) * HDIM, x);
            // ---- cross-attention (single key -> precomputed additive term) ----
            ln_k<<<(int)ROWS, HDIM, 0, stream>>>(x, cadd + (size_t)l * BATCH * HDIM, 2,
                                                 ln_g + (l * 3 + 1) * HDIM, ln_b + (l * 3 + 1) * HDIM, x);
            // ---- FFN ----
            cvt(x, xbf, ROWS * HDIM);
            wmma_gemm_k<<<800, 128, 0, stream>>>(xbf, wb_ff1 + (size_t)l * FFDIM * HDIM,
                                                 ff1_b + l * FFDIM, nullptr, ffh,
                                                 (int)ROWS, FFDIM, HDIM, 1);
            cvt(ffh, xbf, ROWS * FFDIM);
            wmma_gemm_k<<<100, 128, 0, stream>>>(xbf, wb_ff2 + (size_t)l * HDIM * FFDIM,
                                                 ff2_b + l * HDIM, x, tmp,
                                                 (int)ROWS, HDIM, FFDIM, 0);
            ln_k<<<(int)ROWS, HDIM, 0, stream>>>(tmp, nullptr, 0,
                                                 ln_g + (l * 3 + 2) * HDIM, ln_b + (l * 3 + 2) * HDIM, x);
        }

        // ---- vocab projection for position t + greedy argmax ----
        cvt(x, xbf, ROWS * HDIM);
        wmma_logits_k<<<logitsGrid, 128, 0, stream>>>(xbf, wb_fc, fc_b, out, t);
        argmax_k<<<BATCH, 256, 0, stream>>>(out, t, tokens);
    }
}
